// Head_78383153152185
// MI455X (gfx1250) — matrix operand order repaired and checked by reading, not compiled, not run on hardware
//
#include <hip/hip_runtime.h>
#include <math.h>

#define DMODEL 768
#define HS 64
#define NSEQ 4096
#define NBATCH 4
#define ROWS_TOTAL (NBATCH * NSEQ)
#define KTILE 64                      // keys per attention tile
#define NTILES (NSEQ / KTILE)         // 64 tiles per batch

typedef __bf16 bf16_t;
typedef bf16_t bf16x16 __attribute__((ext_vector_type(16)));
typedef float f32x8 __attribute__((ext_vector_type(8)));

union Frag {
    bf16x16 v;
    unsigned short h[16];
};

// native f32 -> bf16 (hardware cvt, RNE)
__device__ __forceinline__ unsigned short f2bf(float f) {
    __bf16 b = (__bf16)f;
    return __builtin_bit_cast(unsigned short, b);
}

__device__ __forceinline__ f32x8 zero8() {
    f32x8 z;
#pragma unroll
    for (int i = 0; i < 8; ++i) z[i] = 0.0f;
    return z;
}

__device__ __forceinline__ f32x8 wmma_bf16(const Frag& a, const Frag& b, f32x8 c) {
    return __builtin_amdgcn_wmma_f32_16x16x32_bf16(
        /*neg_a=*/false, a.v, /*neg_b=*/false, b.v,
        /*c_mod=*/(short)0, c, /*reuse_a=*/false, /*reuse_b=*/false);
}

// ---- gfx1250 async global->LDS path (guarded; falls back to sync copies) ----
#if __has_builtin(__builtin_amdgcn_global_load_async_to_lds_b128)
#define USE_ASYNC_LDS 1
typedef int v4i_vs __attribute__((vector_size(4 * sizeof(int))));
typedef __attribute__((address_space(1))) v4i_vs* gvec1_t;
typedef __attribute__((address_space(3))) v4i_vs* lvec3_t;
#else
#define USE_ASYNC_LDS 0
#endif

__device__ __forceinline__ void copy16_to_lds(const unsigned short* g, unsigned short* l) {
#if USE_ASYNC_LDS
    __builtin_amdgcn_global_load_async_to_lds_b128(
        (gvec1_t)(unsigned short*)g, (lvec3_t)l, 0, 0);
#else
    *(uint4*)l = *(const uint4*)g;
#endif
}

template <int N>
__device__ __forceinline__ void async_wait() {
#if USE_ASYNC_LDS
#if __has_builtin(__builtin_amdgcn_s_wait_asynccnt)
    __builtin_amdgcn_s_wait_asynccnt((unsigned short)N);
#else
    if (N == 0) asm volatile("s_wait_asynccnt 0x0" ::: "memory");
    else        asm volatile("s_wait_asynccnt 0x4" ::: "memory");
#endif
#endif
}

// ---------------------------------------------------------------------------
// Kernel 0 (one-shot): convert W{k,q,v} fp32 -> bf16, transposed [mat][col][768].
// 288 KB table, stays resident in L2, shared by all blocks of kernel 1.
// ---------------------------------------------------------------------------
__global__ __launch_bounds__(256) void wcvt_kernel(
    const float* __restrict__ Wk, const float* __restrict__ Wq,
    const float* __restrict__ Wv, unsigned short* __restrict__ wT)
{
    int idx = blockIdx.x * 256 + threadIdx.x;            // 3*64*768 = 147456
    int mat = idx / (HS * DMODEL);
    int rem = idx - mat * (HS * DMODEL);
    int col = rem / DMODEL;
    int d   = rem - col * DMODEL;
    const float* Wm = (mat == 0) ? Wk : ((mat == 1) ? Wq : Wv);
    wT[idx] = f2bf(Wm[(size_t)d * HS + col]);
}

// ---------------------------------------------------------------------------
// Kernel 1: q/k/v = ix @ W. 128 rows/block, 8 waves x 16 rows.
// ix staged through LDS (coalesced, fp32->bf16); W read pre-converted from L2.
// q scaled by 1/sqrt(64); v written TRANSPOSED per 64-key block: [blk][dim][key].
// ---------------------------------------------------------------------------
__global__ __launch_bounds__(256) void qkv_proj_kernel(
    const float* __restrict__ ix,
    const unsigned short* __restrict__ wT,
    unsigned short* __restrict__ kOut,
    unsigned short* __restrict__ qOut,
    unsigned short* __restrict__ vOut)
{
    __shared__ __attribute__((aligned(16))) unsigned short aT[128 * 32];  // ix tile [row][32]

    const int t = threadIdx.x;
    const int lane = t & 31;
    const int w = t >> 5;
    const int rowBase = blockIdx.x * 128;

    f32x8 acc[3][4];
#pragma unroll
    for (int m = 0; m < 3; ++m)
#pragma unroll
        for (int tt = 0; tt < 4; ++tt)
            acc[m][tt] = zero8();

    for (int kc = 0; kc < DMODEL / 32; ++kc) {
        __syncthreads();
        // ix chunk: 128x32 f32 -> bf16 LDS (1024 float4, 4 per thread, coalesced)
#pragma unroll
        for (int i = 0; i < 4; ++i) {
            int f = t + i * 256;
            int row = f >> 3;
            int d4 = (f & 7) * 4;
            const float4 src = *(const float4*)&ix[(size_t)(rowBase + row) * DMODEL + kc * 32 + d4];
            unsigned short p[4] = {f2bf(src.x), f2bf(src.y), f2bf(src.z), f2bf(src.w)};
            *(unsigned long long*)&aT[row * 32 + d4] = *(const unsigned long long*)p;
        }
        __syncthreads();

        // A fragment: this wave's 16 rows, K=32 chunk
        Frag fa;
        {
            int arow = 16 * w + (lane & 15);
            int kh = (lane >> 4) * 16;
            *(uint4*)&fa.h[0] = *(const uint4*)&aT[arow * 32 + kh];
            *(uint4*)&fa.h[8] = *(const uint4*)&aT[arow * 32 + kh + 8];
        }
#pragma unroll
        for (int m = 0; m < 3; ++m) {
#pragma unroll
            for (int tt = 0; tt < 4; ++tt) {
                Frag fb;  // B from pre-converted transposed W in global (L2-hot)
                const unsigned short* wc =
                    wT + (size_t)m * (HS * DMODEL)
                       + (size_t)(16 * tt + (lane & 15)) * DMODEL
                       + kc * 32 + (lane >> 4) * 16;
                *(uint4*)&fb.h[0] = *(const uint4*)&wc[0];
                *(uint4*)&fb.h[8] = *(const uint4*)&wc[8];
                acc[m][tt] = wmma_bf16(fa, fb, acc[m][tt]);
            }
        }
    }

    // write back; C/D layout: row M = r + 8*(lane>>4), col = 16*tt + (lane&15)
    const int mrow = rowBase + 16 * w + 8 * (lane >> 4);
    const int c0 = lane & 15;
#pragma unroll
    for (int m = 0; m < 3; ++m) {
#pragma unroll
        for (int tt = 0; tt < 4; ++tt)
#pragma unroll
            for (int r = 0; r < 8; ++r) {
                int row = mrow + r;
                int col = 16 * tt + c0;
                float val = acc[m][tt][r];
                if (m == 1) {       // q: fold 1/sqrt(HS)
                    qOut[(size_t)row * HS + col] = f2bf(val * 0.125f);
                } else if (m == 0) {
                    kOut[(size_t)row * HS + col] = f2bf(val);
                } else {            // v: transposed per 64-key block [blk][dim][key]
                    vOut[(size_t)(row >> 6) * (KTILE * HS) + (size_t)col * KTILE + (row & 63)]
                        = f2bf(val);
                }
            }
    }
}

// ---------------------------------------------------------------------------
// Kernel 2: flash attention with double-buffered async global->LDS tiles.
// K tile [key][dim]; V tile already [dim][key] in global (kernel 1); online
// softmax; P layout-bridged through per-wave LDS staging.
// ---------------------------------------------------------------------------
__global__ __launch_bounds__(256) void flash_attn_kernel(
    const unsigned short* __restrict__ qG,
    const unsigned short* __restrict__ kG,
    const unsigned short* __restrict__ vG,
    float* __restrict__ out)
{
    __shared__ __attribute__((aligned(16))) unsigned short kT[2][KTILE * HS];   // [buf][key][dim]
    __shared__ __attribute__((aligned(16))) unsigned short vT[2][KTILE * HS];   // [buf][dim][key]
    __shared__ __attribute__((aligned(16))) unsigned short pS[8 * 16 * 64];     // per-wave P

    const int t = threadIdx.x;
    const int lane = t & 31;
    const int w = t >> 5;
    const int b = blockIdx.x >> 5;          // 32 row-tiles per batch
    const int rowTile = blockIdx.x & 31;
    const int qbase = b * NSEQ + rowTile * 128;

    const unsigned short* kBase = kG + (size_t)b * NSEQ * HS;          // 64 tiles x 8KB
    const unsigned short* vBase = vG + (size_t)b * NTILES * (KTILE * HS);

    // q A-fragments for this wave (K=64 split into two K=32 chunks)
    Frag fqa[2];
    {
        const unsigned short* qr = qG + (size_t)(qbase + 16 * w + (lane & 15)) * HS;
        int kh = (lane >> 4) * 16;
#pragma unroll
        for (int c = 0; c < 2; ++c) {
            *(uint4*)&fqa[c].h[0] = *(const uint4*)&qr[32 * c + kh];
            *(uint4*)&fqa[c].h[8] = *(const uint4*)&qr[32 * c + kh + 8];
        }
    }

    f32x8 o[4];
    float mrow[8], lrow[8];
#pragma unroll
    for (int tt = 0; tt < 4; ++tt) o[tt] = zero8();
#pragma unroll
    for (int r = 0; r < 8; ++r) { mrow[r] = -INFINITY; lrow[r] = 0.0f; }

    unsigned short* pW = &pS[w * 1024];

    // issue one tile-pair: 4 async b128 per thread (K: chunks 0..511, V: 512..1023)
    auto issue_tile = [&](int tile, int buf) {
#pragma unroll
        for (int i = 0; i < 2; ++i) {
            int c = t + i * 256;
            copy16_to_lds(kBase + (size_t)tile * (KTILE * HS) + c * 8, &kT[buf][c * 8]);
            copy16_to_lds(vBase + (size_t)tile * (KTILE * HS) + c * 8, &vT[buf][c * 8]);
        }
    };

    issue_tile(0, 0);   // prologue

    for (int it = 0; it < NTILES; ++it) {
        const int cur = it & 1;
        if (it + 1 < NTILES) {
            issue_tile(it + 1, cur ^ 1);
            async_wait<4>();            // oldest 4 (this tile) complete, prefetch in flight
        } else {
            async_wait<0>();
        }
        __syncthreads();

        const unsigned short* kL = kT[cur];
        const unsigned short* vL = vT[cur];

        // S = q @ K^T  (scale folded into q)
        f32x8 s[4];
#pragma unroll
        for (int tt = 0; tt < 4; ++tt) {
            s[tt] = zero8();
#pragma unroll
            for (int c = 0; c < 2; ++c) {
                Frag fb;
                int off = (16 * tt + (lane & 15)) * 64 + 32 * c + (lane >> 4) * 16;
                *(uint4*)&fb.h[0] = *(const uint4*)&kL[off];
                *(uint4*)&fb.h[8] = *(const uint4*)&kL[off + 8];
                s[tt] = wmma_bf16(fqa[c], fb, s[tt]);
            }
        }

        // online softmax; VGPR slot r holds one row per 16-lane half
#pragma unroll
        for (int r = 0; r < 8; ++r) {
            float mloc = fmaxf(fmaxf(s[0][r], s[1][r]), fmaxf(s[2][r], s[3][r]));
            mloc = fmaxf(mloc, __shfl_xor(mloc, 1, 32));
            mloc = fmaxf(mloc, __shfl_xor(mloc, 2, 32));
            mloc = fmaxf(mloc, __shfl_xor(mloc, 4, 32));
            mloc = fmaxf(mloc, __shfl_xor(mloc, 8, 32));
            float mnew = fmaxf(mrow[r], mloc);
            float alpha = __expf(mrow[r] - mnew);
            mrow[r] = mnew;
            float psum = 0.0f;
#pragma unroll
            for (int tt = 0; tt < 4; ++tt) {
                float p = __expf(s[tt][r] - mnew);
                s[tt][r] = p;
                psum += p;
            }
            psum += __shfl_xor(psum, 1, 32);
            psum += __shfl_xor(psum, 2, 32);
            psum += __shfl_xor(psum, 4, 32);
            psum += __shfl_xor(psum, 8, 32);
            lrow[r] = lrow[r] * alpha + psum;
#pragma unroll
            for (int tt = 0; tt < 4; ++tt) o[tt][r] *= alpha;
        }

        // stage P (bf16) to LDS: C/D layout -> A-fragment layout bridge
        {
            int mOff = 8 * (lane >> 4);
            int c0 = lane & 15;
#pragma unroll
            for (int tt = 0; tt < 4; ++tt)
#pragma unroll
                for (int r = 0; r < 8; ++r)
                    pW[(r + mOff) * 64 + 16 * tt + c0] = f2bf(s[tt][r]);
        }

        // O += P @ V  (V tile already [dim][key] -> contiguous B-fragments)
#pragma unroll
        for (int c = 0; c < 2; ++c) {
            Frag fpa;
            {
                int prow = (lane & 15) * 64;
                int kh = (lane >> 4) * 16;
                *(uint4*)&fpa.h[0] = *(const uint4*)&pW[prow + 32 * c + kh];
                *(uint4*)&fpa.h[8] = *(const uint4*)&pW[prow + 32 * c + kh + 8];
            }
#pragma unroll
            for (int tt = 0; tt < 4; ++tt) {
                Frag fb;
                int off = (16 * tt + (lane & 15)) * 64 + 32 * c + (lane >> 4) * 16;
                *(uint4*)&fb.h[0] = *(const uint4*)&vL[off];
                *(uint4*)&fb.h[8] = *(const uint4*)&vL[off + 8];
                o[tt] = wmma_bf16(fpa, fb, o[tt]);
            }
        }
        __syncthreads();   // all waves done with buf[cur] before it gets re-filled
    }

    // normalize and write fp32 output
    {
        int mrowg = qbase + 16 * w + 8 * (lane >> 4);
        int c0 = lane & 15;
#pragma unroll
        for (int tt = 0; tt < 4; ++tt)
#pragma unroll
            for (int r = 0; r < 8; ++r)
                out[(size_t)(mrowg + r) * HS + 16 * tt + c0] = o[tt][r] / lrow[r];
    }
}

extern "C" void kernel_launch(void* const* d_in, const int* in_sizes, int n_in,
                              void* d_out, int out_size, void* d_ws, size_t ws_size,
                              hipStream_t stream) {
    const float* ix = (const float*)d_in[0];
    const float* Wk = (const float*)d_in[1];
    const float* Wq = (const float*)d_in[2];
    const float* Wv = (const float*)d_in[3];

    unsigned short* ws = (unsigned short*)d_ws;
    unsigned short* kB = ws;                                   // [row][64]
    unsigned short* qB = kB + (size_t)ROWS_TOTAL * HS;         // [row][64], pre-scaled
    unsigned short* vB = qB + (size_t)ROWS_TOTAL * HS;         // [blk][dim][key]
    unsigned short* wTb = vB + (size_t)ROWS_TOTAL * HS;        // [mat][col][768]

    wcvt_kernel<<<(3 * HS * DMODEL) / 256, 256, 0, stream>>>(Wk, Wq, Wv, wTb);
    qkv_proj_kernel<<<ROWS_TOTAL / 128, 256, 0, stream>>>(ix, wTb, kB, qB, vB);
    flash_attn_kernel<<<ROWS_TOTAL / 128, 256, 0, stream>>>(qB, kB, vB, (float*)d_out);
}
